// RealQuantizedLinearGroupWise_4518305595686
// MI455X (gfx1250) — matrix operand order repaired and checked by reading, not compiled, not run on hardware
//
#include <hip/hip_runtime.h>

typedef __attribute__((ext_vector_type(16))) _Float16 v16h;
typedef __attribute__((ext_vector_type(8)))  _Float16 v8h;
typedef __attribute__((ext_vector_type(8)))  float    v8f;
typedef __attribute__((ext_vector_type(4)))  int      v4i;

typedef __attribute__((address_space(1))) v4i* gp_v4i;  // global int4*
typedef __attribute__((address_space(3))) v4i* lp_v4i;  // LDS int4*

#define BM 128
#define BN 128
#define BK 64
#define LDK 72            // padded LDS row stride in halves (+16B)
#define KDIM 4096
#define NDIM 11008
#define MDIM 8192
#define NT (KDIM / BK)    // 64 K-tiles
#define NGRP (KDIM / 128) // 32 scale groups

#if __has_builtin(__builtin_amdgcn_global_load_async_to_lds_b128)
#define HAVE_ASYNC_LDS 1
#else
#define HAVE_ASYNC_LDS 0
#endif

__device__ __forceinline__ void wait_async0() {
#if __has_builtin(__builtin_amdgcn_s_wait_asynccnt)
  __builtin_amdgcn_s_wait_asynccnt(0);
#else
  asm volatile("s_wait_asynccnt 0" ::: "memory");
#endif
}

__device__ __forceinline__ void async_g2l_b128(const _Float16* g, _Float16* l) {
#if HAVE_ASYNC_LDS
  __builtin_amdgcn_global_load_async_to_lds_b128((gp_v4i)(void*)g, (lp_v4i)l,
                                                 0, 0);
#else
  *(v8h*)l = *(const v8h*)g;  // sync fallback: global_load_b128 + ds_store_b128
#endif
}

union AF { v16h v; v8h p[2]; };
union H8 { uint4 u; _Float16 h[8]; };

__global__ __launch_bounds__(256) void qlinear_wmma_f16(
    const _Float16* __restrict__ x, const int* __restrict__ w,
    const _Float16* __restrict__ sc, const _Float16* __restrict__ bs,
    _Float16* __restrict__ out) {
  __shared__ __align__(16) _Float16 sA[2][BM * LDK];
  __shared__ __align__(16) _Float16 sB[2][BN * LDK];

  const int tid = threadIdx.x;
  const int bx = blockIdx.x;  // N tile 0..85
  const int by = blockIdx.y;  // M tile 0..63

  // ---- copy roles: 2 threads per row, 32 halves / 32 ints each
  const int crow = tid >> 1;         // 0..127
  const int ckof = (tid & 1) * 32;   // 0 or 32
  const _Float16* xg = x + (size_t)(by * BM + crow) * KDIM + ckof;
  const int*      wg = w + (size_t)(bx * BN + crow) * KDIM + ckof;
  const _Float16* sg = sc + (size_t)(bx * BN + crow) * NGRP;

  // ---- wave roles
  const int wave = tid >> 5;
  const int lane = tid & 31;
  const int wm = (wave & 3) * 32;    // 0..96
  const int wn = (wave >> 2) * 64;   // 0 or 64
  const int lr = lane & 15;
  const int lh = lane >> 4;

  v8f acc[2][4] = {};
  int4 wreg[8];

  // ---------------- helpers as lambdas ----------------
  auto loadA = [&](int kt, int buf) {
#pragma unroll
    for (int j = 0; j < 4; ++j)
      async_g2l_b128(xg + (size_t)kt * BK + j * 8,
                     &sA[buf][crow * LDK + ckof + j * 8]);
  };
  auto loadW = [&](int kt) {
    const int4* p = (const int4*)(wg + (size_t)kt * BK);
#pragma unroll
    for (int j = 0; j < 8; ++j) wreg[j] = p[j];
    if (kt + 2 < NT)  // stream prefetch 2 tiles ahead -> global_prefetch_b8
      __builtin_prefetch(wg + (size_t)(kt + 2) * BK, 0, 1);
  };
  auto storeW = [&](int kt, int buf) {
    // int8 values are exact in fp16; reference multiplies f16(w) * f16(scale)
    // in fp16 -> use v_cvt_f16_i16 + (packed) f16 multiplies, no f32 detour.
    const _Float16 hs = sg[kt >> 1];  // one scale per (row, 128-group)
#pragma unroll
    for (int j = 0; j < 4; ++j) {
      H8 o;
      const int* a = (const int*)&wreg[2 * j];
#pragma unroll
      for (int e = 0; e < 8; ++e)
        o.h[e] = (_Float16)(short)a[e] * hs;
      *(uint4*)&sB[buf][crow * LDK + ckof + j * 8] = o.u;
    }
  };
  auto compute = [&](int buf) {
#pragma unroll
    for (int kk = 0; kk < BK; kk += 32) {
      AF aF[2], bF[4];
#pragma unroll
      for (int i = 0; i < 2; ++i) {  // A 16x32: lane<16 -> K[0..7],[16..23]
        const int row = wm + i * 16 + lr;
        const int c0 = kk + lh * 8;
        aF[i].p[0] = *(const v8h*)&sA[buf][row * LDK + c0];
        aF[i].p[1] = *(const v8h*)&sA[buf][row * LDK + c0 + 16];
      }
#pragma unroll
      for (int j = 0; j < 4; ++j) {  // B 32x16: lane<16 -> K[0..7],[16..23]
        const int col = wn + j * 16 + lr;
        const int c0 = kk + lh * 8;
        bF[j].p[0] = *(const v8h*)&sB[buf][col * LDK + c0];
        bF[j].p[1] = *(const v8h*)&sB[buf][col * LDK + c0 + 16];
      }
#pragma unroll
      for (int i = 0; i < 2; ++i)
#pragma unroll
        for (int j = 0; j < 4; ++j)
          acc[i][j] = __builtin_amdgcn_wmma_f32_16x16x32_f16(
              false, aF[i].v, false, bF[j].v, (short)0, acc[i][j], false, false);
    }
  };

  // ---------------- pipeline ----------------
  loadA(0, 0);
  loadW(0);
  storeW(0, 0);
#if HAVE_ASYNC_LDS
  wait_async0();
#endif
  __syncthreads();

  int buf = 0;
  for (int kt = 0; kt < NT; ++kt) {
    const bool pf = (kt + 1 < NT);
    if (pf) { loadA(kt + 1, buf ^ 1); loadW(kt + 1); }
    compute(buf);
    if (pf) {
      storeW(kt + 1, buf ^ 1);
#if HAVE_ASYNC_LDS
      wait_async0();
#endif
    }
    __syncthreads();
    buf ^= 1;
  }

  // ---------------- epilogue: bias + fp16 store ----------------
#pragma unroll
  for (int i = 0; i < 2; ++i) {
#pragma unroll
    for (int j = 0; j < 4; ++j) {
      const int n = bx * BN + wn + j * 16 + lr;
      const int m0 = by * BM + wm + i * 16 + lh * 8;
      const float bf = (float)bs[n];
      _Float16* o = out + (size_t)m0 * NDIM + n;
#pragma unroll
      for (int r = 0; r < 8; ++r)
        o[(size_t)r * NDIM] = (_Float16)(acc[i][j][r] + bf);
    }
  }
}

extern "C" void kernel_launch(void* const* d_in, const int* in_sizes, int n_in,
                              void* d_out, int out_size, void* d_ws, size_t ws_size,
                              hipStream_t stream) {
  const _Float16* x = (const _Float16*)d_in[0];
  const int* wq = (const int*)d_in[1];
  const _Float16* sc = (const _Float16*)d_in[2];
  const _Float16* bs = (const _Float16*)d_in[3];
  _Float16* out = (_Float16*)d_out;

  dim3 grid(NDIM / BN, MDIM / BM);  // 86 x 64
  qlinear_wmma_f16<<<grid, dim3(256), 0, stream>>>(x, wq, sc, bs, out);
}
